// Update_46196668235890
// MI455X (gfx1250) — matrix operand order repaired and checked by reading, not compiled, not run on hardware
//
#include <hip/hip_runtime.h>
#include <cstdint>
#include <cstddef>

#define DIM      384
#define CORR_K   882
#define CORR_KP  896
#define CAP_KK   4096
#define CAP_IJ   576
#define HN_SIZE  (1u<<18)
#define HK_SIZE  (1u<<14)
#define HI_SIZE  (1u<<11)
#define LNEPS    1e-3f
#define NEG_INF_KEY 0x807FFFFFu

typedef unsigned long long ull;
typedef __attribute__((ext_vector_type(16))) __bf16 v16bf;
typedef __attribute__((ext_vector_type(8)))  float  v8f;

union FragAB { v16bf v; uint4 q[2]; };

__device__ __forceinline__ unsigned short f2bf(float f) {
  unsigned int b = __float_as_uint(f);
  b += 0x7FFFu + ((b >> 16) & 1u);          // round-to-nearest-even
  return (unsigned short)(b >> 16);
}
__device__ __forceinline__ int fkey(float f) {
  int b = __float_as_int(f);
  return b >= 0 ? b : (b ^ 0x7FFFFFFF);     // order-preserving float->int
}
__device__ __forceinline__ float fkey_dec(int k) {
  return __int_as_float(k >= 0 ? k : (k ^ 0x7FFFFFFF));
}
__device__ __forceinline__ unsigned int hmix(ull key, unsigned int mask) {
  key *= 0x9E3779B97F4A7C15ull;
  return (unsigned int)(key >> 32) & mask;
}

// ---------------------------------------------------------------------------
// WMMA GEMM:  C[M,384] = act( gather(A)[M,K] @ Wt_bf16[384][Kpad] + bias (+resid) )
// 128x64 macro-tile, 8 waves x (2x2) 16x16 sub-tiles, K-step 32.
// bf16 operands / f32 accumulate via v_wmma_f32_16x16x32_bf16.
// AL4: A rows are 16B-aligned and K%4==0 (float4 path, no tail).
// act: 0=none 1=relu 2=sigmoid.  gidx: optional row gather (neg -> zero row).
// ---------------------------------------------------------------------------
template <bool AL4>
__global__ __launch_bounds__(256) void k_gemm(
    const float* __restrict__ A, const unsigned short* __restrict__ Wt,
    const float* __restrict__ bias, const float* resid,
    const int* __restrict__ gidx, float* C,
    int M, int K, int Kpad, int act)
{
  __shared__ unsigned short lA[128 * 40];   // [m][k] bf16, padded stride 40
  __shared__ unsigned short lB[64 * 40];    // [n][k] bf16, padded stride 40

  const int t  = threadIdx.x;
  const int bm = blockIdx.x * 128;
  const int bn = blockIdx.y * 64;

  // A staging: 2 threads per row, 16 consecutive k each
  const int ar = t >> 1;
  const int ac = (t & 1) * 16;
  const int gm = bm + ar;
  bool avalid = (gm < M);
  const float* arow = A;
  if (avalid) {
    if (gidx) { int s = gidx[gm]; if (s < 0) avalid = false; else arow = A + (long long)s * K; }
    else arow = A + (long long)gm * K;
  }

  // B staging: 4 threads per n-row, 8 consecutive k each (pre-transposed weights)
  const int bnn = t >> 2;
  const int bk8 = (t & 3) * 8;
  const unsigned short* wrow = Wt + (long long)(bn + bnn) * Kpad + bk8;

  const int wv = t >> 5, lane = t & 31;
  const int mg  = (wv & 3) * 32;       // wave's 32-row m group
  const int ng  = (wv >> 2) * 32;      // wave's 32-col n group
  const int l15 = lane & 15, lhi = lane >> 4;

  v8f acc[2][2];
  for (int a = 0; a < 2; ++a)
    for (int b = 0; b < 2; ++b)
      for (int i = 0; i < 8; ++i) acc[a][b][i] = 0.f;

  const int nk = Kpad >> 5;
  for (int kt = 0; kt < nk; ++kt) {
    const int k0 = kt << 5;
    { // ---- stage A (f32 -> bf16, b128 LDS stores) ----
      float v[16];
      if (!avalid) {
        for (int i = 0; i < 16; ++i) v[i] = 0.f;
      } else if (AL4) {
        const float4* p4 = (const float4*)(arow + k0 + ac);
        float4 f0 = p4[0], f1 = p4[1], f2 = p4[2], f3 = p4[3];
        v[0]=f0.x; v[1]=f0.y; v[2]=f0.z;  v[3]=f0.w;
        v[4]=f1.x; v[5]=f1.y; v[6]=f1.z;  v[7]=f1.w;
        v[8]=f2.x; v[9]=f2.y; v[10]=f2.z; v[11]=f2.w;
        v[12]=f3.x; v[13]=f3.y; v[14]=f3.z; v[15]=f3.w;
      } else if (k0 + 32 <= K) {        // full tile, rows 8B-aligned (K even)
        const float2* p2 = (const float2*)(arow + k0 + ac);
        for (int j = 0; j < 8; ++j) { float2 f = p2[j]; v[2*j] = f.x; v[2*j+1] = f.y; }
      } else {                          // single guarded tail tile
        for (int i = 0; i < 16; ++i) { int k = k0 + ac + i; v[i] = (k < K) ? arow[k] : 0.f; }
      }
      union { uint4 q; unsigned short s[8]; } u0, u1;
      for (int i = 0; i < 8; ++i) { u0.s[i] = f2bf(v[i]); u1.s[i] = f2bf(v[8 + i]); }
      *(uint4*)&lA[ar * 40 + ac]     = u0.q;
      *(uint4*)&lA[ar * 40 + ac + 8] = u1.q;
    }
    { // ---- stage B: straight b128 copy ----
      *(uint4*)&lB[bnn * 40 + bk8] = *(const uint4*)(wrow + k0);
    }
    __syncthreads();

    FragAB fa[2], fb[2];
    const int kh = lhi * 8, kb = lhi * 16;
    for (int im = 0; im < 2; ++im) {    // A fragment per ISA 16-bit 16x32 layout
      int m = mg + im * 16 + l15;
      fa[im].q[0] = *(const uint4*)&lA[m * 40 + kh];
      fa[im].q[1] = *(const uint4*)&lA[m * 40 + 16 + kh];
    }
    for (int in = 0; in < 2; ++in) {    // B fragment per ISA 16-bit 32x16 layout (same k order as A)
      int n = ng + in * 16 + l15;
      fb[in].q[0] = *(const uint4*)&lB[n * 40 + kh];
      fb[in].q[1] = *(const uint4*)&lB[n * 40 + 16 + kh];
    }
    for (int im = 0; im < 2; ++im)
      for (int in = 0; in < 2; ++in)
        acc[im][in] = __builtin_amdgcn_wmma_f32_16x16x32_bf16(
            false, fa[im].v, false, fb[in].v, (short)0, acc[im][in], false, false);
    __syncthreads();
  }

  for (int im = 0; im < 2; ++im)
    for (int in = 0; in < 2; ++in) {
      int gn = bn + ng + in * 16 + l15;
      float bv = bias[gn];
      for (int v = 0; v < 8; ++v) {
        int gmo = bm + mg + im * 16 + lhi * 8 + v;
        if (gmo >= M) continue;
        float r = acc[im][in][v] + bv;
        if (resid) r += resid[(long long)gmo * DIM + gn];
        if (act == 1) r = fmaxf(r, 0.f);
        else if (act == 2) r = 1.f / (1.f + __expf(-r));
        C[(long long)gmo * DIM + gn] = r;
      }
    }
}

// ---------------------------------------------------------------------------
// LayerNorm: one wave32 per 384-wide row, optional fused ReLU. In-place safe.
// ---------------------------------------------------------------------------
__global__ __launch_bounds__(256) void k_ln(const float* X, const float* __restrict__ g,
                                            const float* __restrict__ b, float* Y,
                                            int dorelu, int M)
{
  int row = blockIdx.x * 8 + (threadIdx.x >> 5);
  if (row >= M) return;
  int lane = threadIdx.x & 31;
  const float* x = X + (long long)row * DIM;
  float v[12]; float s = 0.f;
  for (int i = 0; i < 12; ++i) { v[i] = x[lane + 32 * i]; s += v[i]; }
  for (int o = 16; o > 0; o >>= 1) s += __shfl_xor(s, o, 32);
  float mean = s * (1.f / DIM);
  float q = 0.f;
  for (int i = 0; i < 12; ++i) { float d = v[i] - mean; q += d * d; }
  for (int o = 16; o > 0; o >>= 1) q += __shfl_xor(q, o, 32);
  float inv = rsqrtf(q * (1.f / DIM) + LNEPS);
  float* y = Y + (long long)row * DIM;
  for (int i = 0; i < 12; ++i) {
    int c = lane + 32 * i;
    float r = (v[i] - mean) * inv * g[c] + b[c];
    if (dorelu) r = fmaxf(r, 0.f);
    y[c] = r;
  }
}

// -------------------------- small utility kernels --------------------------
__global__ void k_fill32(unsigned int* p, unsigned int v, int n) {
  int i = blockIdx.x * 256 + threadIdx.x; if (i < n) p[i] = v;
}
__global__ void k_fill64(ull* p, ull v, int n) {
  int i = blockIdx.x * 256 + threadIdx.x; if (i < n) p[i] = v;
}
__global__ void k_add2(const float* a, const float* b, float* o, int n) {
  int i = blockIdx.x * 256 + threadIdx.x; if (i < n) o[i] = a[i] + b[i];
}
__global__ void k_fma3(const float* x, const float* g, const float* r, float* o, int n) {
  int i = blockIdx.x * 256 + threadIdx.x; if (i < n) o[i] = x[i] + g[i] * r[i];
}
// transpose + convert: O[n][k] (bf16, K zero-padded to Kpad) from W[k][n] (f32)
__global__ void k_convw(const float* __restrict__ W, unsigned short* __restrict__ O,
                        int K, int N, int Kpad, int total) {
  int i = blockIdx.x * 256 + threadIdx.x; if (i >= total) return;
  int n = i / Kpad, k = i - n * Kpad;
  O[i] = (k < K) ? f2bf(W[(long long)k * N + n]) : (unsigned short)0;
}

// ------------------------------ hash kernels -------------------------------
__global__ void k_hash_insert(const int* __restrict__ p1, long long m1,
                              const int* __restrict__ p2,
                              ull* keys, int* vals, unsigned int mask, int E)
{
  int e = blockIdx.x * 256 + threadIdx.x; if (e >= E) return;
  long long key = (long long)p1[e] * m1 + (p2 ? (long long)p2[e] : 0ll);
  ull uk = (ull)key;
  unsigned int h = hmix(uk, mask);
  for (;;) {
    ull old = atomicCAS(&keys[h], ~0ull, uk);
    if (old == ~0ull) { if (vals) vals[h] = e; break; }
    if (old == uk) break;
    h = (h + 1) & mask;
  }
}
__device__ __forceinline__ int h_find(const ull* keys, const int* vals,
                                      unsigned int mask, long long key) {
  ull uk = (ull)key;
  unsigned int h = hmix(uk, mask);
  for (unsigned int i = 0; i <= mask; ++i) {
    ull k = keys[h];
    if (k == uk) return vals[h];
    if (k == ~0ull) return -1;
    h = (h + 1) & mask;
  }
  return -1;
}
__global__ void k_neigh(const int* __restrict__ kk, const int* __restrict__ jj,
                        const ull* __restrict__ keys, const int* __restrict__ vals,
                        int* ix, int* jx, int E)
{
  int e = blockIdx.x * 256 + threadIdx.x; if (e >= E) return;
  long long key = (long long)kk[e] * 65536ll + (long long)jj[e];
  ix[e] = h_find(keys, vals, HN_SIZE - 1, key - 1);
  jx[e] = h_find(keys, vals, HN_SIZE - 1, key + 1);
}
__global__ void k_group_assign(const ull* __restrict__ keys, int* ids,
                               int* counter, unsigned int tsize, int cap)
{
  int s = blockIdx.x * 256 + threadIdx.x; if (s >= (int)tsize) return;
  if (keys[s] != ~0ull) {
    int id = atomicAdd(counter, 1);
    ids[s] = id < cap ? id : (cap - 1);
  }
}
__global__ void k_group_get(const int* __restrict__ p1, long long m1,
                            const int* __restrict__ p2,
                            const ull* __restrict__ keys, const int* __restrict__ ids,
                            unsigned int mask, int* gid, int E)
{
  int e = blockIdx.x * 256 + threadIdx.x; if (e >= E) return;
  long long key = (long long)p1[e] * m1 + (p2 ? (long long)p2[e] : 0ll);
  ull uk = (ull)key;
  unsigned int h = hmix(uk, mask);
  for (;;) { if (keys[h] == uk) { gid[e] = ids[h]; return; } h = (h + 1) & mask; }
}

// ------------------------- segment softmax kernels -------------------------
__global__ void k_segmax(const float* __restrict__ GX, const int* __restrict__ gid,
                         int* GMAX, int n) {
  int i = blockIdx.x * 256 + threadIdx.x; if (i >= n) return;
  int e = i / DIM, d = i - e * DIM;
  atomicMax(&GMAX[gid[e] * DIM + d], fkey(GX[i]));
}
__global__ void k_gedens(float* GX, const int* __restrict__ gid,
                         const int* __restrict__ GMAX, float* DEN, int n) {
  int i = blockIdx.x * 256 + threadIdx.x; if (i >= n) return;
  int e = i / DIM, d = i - e * DIM;
  float m  = fkey_dec(GMAX[gid[e] * DIM + d]);
  float ge = __expf(GX[i] - m);
  GX[i] = ge;
  atomicAdd(&DEN[gid[e] * DIM + d], ge);
}
__global__ void k_yacc(const float* __restrict__ FX, const float* __restrict__ GE,
                       const int* __restrict__ gid, const float* __restrict__ DEN,
                       float* Y, int n) {
  int i = blockIdx.x * 256 + threadIdx.x; if (i >= n) return;
  int e = i / DIM, d = i - e * DIM;
  int gi = gid[e] * DIM + d;
  atomicAdd(&Y[gi], FX[i] * GE[i] / DEN[gi]);
}
__global__ void k_scatadd(float* X, const float* __restrict__ H,
                          const int* __restrict__ gid, int n) {
  int i = blockIdx.x * 256 + threadIdx.x; if (i >= n) return;
  int e = i / DIM, d = i - e * DIM;
  X[i] += H[gid[e] * DIM + d];
}

// --------------------------------- heads -----------------------------------
__global__ __launch_bounds__(256) void k_heads(const float* __restrict__ X,
    const float* __restrict__ wd, const float* __restrict__ bd,
    const float* __restrict__ ww, const float* __restrict__ bw,
    float* dout, float* wout, int E)
{
  int row = blockIdx.x * 8 + (threadIdx.x >> 5);
  if (row >= E) return;
  int lane = threadIdx.x & 31;
  const float* x = X + (long long)row * DIM;
  float d0 = 0, d1 = 0, w0 = 0, w1 = 0;
  for (int c = lane; c < DIM; c += 32) {
    float v = fmaxf(x[c], 0.f);
    d0 += v * wd[c * 2]; d1 += v * wd[c * 2 + 1];
    w0 += v * ww[c * 2]; w1 += v * ww[c * 2 + 1];
  }
  for (int o = 16; o > 0; o >>= 1) {
    d0 += __shfl_xor(d0, o, 32); d1 += __shfl_xor(d1, o, 32);
    w0 += __shfl_xor(w0, o, 32); w1 += __shfl_xor(w1, o, 32);
  }
  if (lane == 0) {
    dout[row * 2]     = d0 + bd[0];
    dout[row * 2 + 1] = d1 + bd[1];
    wout[row * 2]     = 1.f / (1.f + __expf(-(w0 + bw[0])));
    wout[row * 2 + 1] = 1.f / (1.f + __expf(-(w1 + bw[1])));
  }
}

// ---------------------------------------------------------------------------
// d_in layout (pytree leaves, dict keys sorted alphabetically, b before w)
// ---------------------------------------------------------------------------
enum {
  IN_NET = 0, IN_INP, IN_CORR, IN_FLOW, IN_II, IN_JJ, IN_KK,
  P_AIJ_F_B, P_AIJ_F_W, P_AIJ_G_B, P_AIJ_G_W, P_AIJ_H_B, P_AIJ_H_W,
  P_AKK_F_B, P_AKK_F_W, P_AKK_G_B, P_AKK_G_W, P_AKK_H_B, P_AKK_H_W,
  P_C1_L1_B, P_C1_L1_W, P_C1_L2_B, P_C1_L2_W,
  P_C2_L1_B, P_C2_L1_W, P_C2_L2_B, P_C2_L2_W,
  P_CO_L1_B, P_CO_L1_W, P_CO_L2_B, P_CO_L2_W, P_CO_L3_B, P_CO_L3_W,
  P_CO_LN_B, P_CO_LN_G,
  P_D_B, P_D_W,
  P_G1_GATE_B, P_G1_GATE_W, P_G1_R1_B, P_G1_R1_W, P_G1_R2_B, P_G1_R2_W,
  P_G2_GATE_B, P_G2_GATE_W, P_G2_R1_B, P_G2_R1_W, P_G2_R2_B, P_G2_R2_W,
  P_LN1_B, P_LN1_G, P_LN2_B, P_LN2_G,
  P_NORM_B, P_NORM_G,
  P_W_B, P_W_W
};

extern "C" void kernel_launch(void* const* d_in, const int* in_sizes, int n_in,
                              void* d_out, int out_size, void* d_ws, size_t ws_size,
                              hipStream_t stream)
{
  const float* net  = (const float*)d_in[IN_NET];
  const float* inp  = (const float*)d_in[IN_INP];
  const float* corr = (const float*)d_in[IN_CORR];
  const int*   ii   = (const int*)d_in[IN_II];
  const int*   jj   = (const int*)d_in[IN_JJ];
  const int*   kk   = (const int*)d_in[IN_KK];
  const int E  = in_sizes[IN_II];
  const int nE = E * DIM;

  // ---- workspace carve-out (256B aligned) ----
  char* p = (char*)d_ws;
  auto alloc = [&](size_t b) -> void* { void* r = (void*)p; p += (b + 255) & ~(size_t)255; return r; };

  unsigned short* Wco1 = (unsigned short*)alloc((size_t)CORR_KP * DIM * 2);
  unsigned short* Wm[18];
  for (int i = 0; i < 18; ++i) Wm[i] = (unsigned short*)alloc((size_t)DIM * DIM * 2);
  // order: CO2,CO3, C1A,C1B, C2A,C2B, KKF,KKG,KKH, IJF,IJG,IJH, G1GATE,G1R1,G1R2, G2GATE,G2R1,G2R2
  float* B0 = (float*)alloc((size_t)nE * 4);
  float* B1 = (float*)alloc((size_t)nE * 4);
  float* B2 = (float*)alloc((size_t)nE * 4);
  float* B3 = (float*)alloc((size_t)nE * 4);
  ull* hnK = (ull*)alloc((size_t)HN_SIZE * 8);  int* hnV = (int*)alloc((size_t)HN_SIZE * 4);
  ull* hkK = (ull*)alloc((size_t)HK_SIZE * 8);  int* hkI = (int*)alloc((size_t)HK_SIZE * 4);
  ull* hiK = (ull*)alloc((size_t)HI_SIZE * 8);  int* hiI = (int*)alloc((size_t)HI_SIZE * 4);
  int* counters = (int*)alloc(8);
  int* ix   = (int*)alloc((size_t)E * 4);
  int* jx   = (int*)alloc((size_t)E * 4);
  int* gidk = (int*)alloc((size_t)E * 4);
  int* gidi = (int*)alloc((size_t)E * 4);
  int*   GMk = (int*)alloc((size_t)CAP_KK * DIM * 4);
  float* DNk = (float*)alloc((size_t)CAP_KK * DIM * 4);
  float* Yk  = (float*)alloc((size_t)CAP_KK * DIM * 4);
  float* Hk  = (float*)alloc((size_t)CAP_KK * DIM * 4);
  int*   GMi = (int*)alloc((size_t)CAP_IJ * DIM * 4);
  float* DNi = (float*)alloc((size_t)CAP_IJ * DIM * 4);
  float* Yi  = (float*)alloc((size_t)CAP_IJ * DIM * 4);
  float* Hi  = (float*)alloc((size_t)CAP_IJ * DIM * 4);
  if ((size_t)(p - (char*)d_ws) > ws_size) return;  // workspace too small

  float* xo = (float*)d_out;                 // x: [E,384]
  float* dd = xo + (size_t)E * DIM;          // d: [E,2]
  float* wo = dd + (size_t)E * 2;            // w: [E,2]

  const dim3 blk(256);
  auto ew = [](int n) { return dim3((unsigned)((n + 255) / 256)); };
  auto gemm = [&](const float* A, const unsigned short* W, int biasIdx,
                  const float* resid, const int* gx, float* C,
                  int M, int K, int Kpad, int act) {
    dim3 grid((unsigned)((M + 127) / 128), DIM / 64);
    if ((K & 3) == 0)
      k_gemm<true><<<grid, blk, 0, stream>>>(A, W, (const float*)d_in[biasIdx],
                                             resid, gx, C, M, K, Kpad, act);
    else
      k_gemm<false><<<grid, blk, 0, stream>>>(A, W, (const float*)d_in[biasIdx],
                                              resid, gx, C, M, K, Kpad, act);
  };
  auto conv = [&](int wIdx, unsigned short* dst, int K, int Kpad) {
    int tot = DIM * Kpad;   // output is [N=384][Kpad]
    k_convw<<<ew(tot), blk, 0, stream>>>((const float*)d_in[wIdx], dst, K, DIM, Kpad, tot);
  };

  // ---- convert all weights to bf16, transposed [n][k], K zero-padded ----
  conv(P_CO_L1_W, Wco1, CORR_K, CORR_KP);
  conv(P_CO_L2_W,   Wm[0],  DIM, DIM); conv(P_CO_L3_W,   Wm[1],  DIM, DIM);
  conv(P_C1_L1_W,   Wm[2],  DIM, DIM); conv(P_C1_L2_W,   Wm[3],  DIM, DIM);
  conv(P_C2_L1_W,   Wm[4],  DIM, DIM); conv(P_C2_L2_W,   Wm[5],  DIM, DIM);
  conv(P_AKK_F_W,   Wm[6],  DIM, DIM); conv(P_AKK_G_W,   Wm[7],  DIM, DIM);
  conv(P_AKK_H_W,   Wm[8],  DIM, DIM);
  conv(P_AIJ_F_W,   Wm[9],  DIM, DIM); conv(P_AIJ_G_W,   Wm[10], DIM, DIM);
  conv(P_AIJ_H_W,   Wm[11], DIM, DIM);
  conv(P_G1_GATE_W, Wm[12], DIM, DIM); conv(P_G1_R1_W,   Wm[13], DIM, DIM);
  conv(P_G1_R2_W,   Wm[14], DIM, DIM);
  conv(P_G2_GATE_W, Wm[15], DIM, DIM); conv(P_G2_R1_W,   Wm[16], DIM, DIM);
  conv(P_G2_R2_W,   Wm[17], DIM, DIM);

  // ---- corr encoder + input sum + norm ----
  k_add2<<<ew(nE), blk, 0, stream>>>(net, inp, B1, nE);                         // S = net+inp
  gemm(corr, Wco1, P_CO_L1_B, nullptr, nullptr, B2, E, CORR_K, CORR_KP, 1);     // T1 = relu(corr@W1+b)
  gemm(B2, Wm[0], P_CO_L2_B, nullptr, nullptr, B3, E, DIM, DIM, 0);             // C2
  k_ln<<<ew(E * 32), blk, 0, stream>>>(B3, (const float*)d_in[P_CO_LN_G],
      (const float*)d_in[P_CO_LN_B], B2, 1, E);                                 // T2 = relu(LN(C2))
  gemm(B2, Wm[1], P_CO_L3_B, B1, nullptr, B0, E, DIM, DIM, 0);                  // X = T2@W3+b + S
  k_ln<<<ew(E * 32), blk, 0, stream>>>(B0, (const float*)d_in[P_NORM_G],
      (const float*)d_in[P_NORM_B], B0, 0, E);                                  // X0 = LN(X)

  // ---- neighbors via hash on kk*65536+jj ----
  k_fill64<<<ew(HN_SIZE), blk, 0, stream>>>(hnK, ~0ull, HN_SIZE);
  k_hash_insert<<<ew(E), blk, 0, stream>>>(kk, 65536ll, jj, hnK, hnV, HN_SIZE - 1, E);
  k_neigh<<<ew(E), blk, 0, stream>>>(kk, jj, hnK, hnV, ix, jx, E);

  // ---- temporal convs c1 / c2 (gathered rows, zero-masked) ----
  gemm(B0, Wm[2], P_C1_L1_B, nullptr, ix, B1, E, DIM, DIM, 1);
  gemm(B1, Wm[3], P_C1_L2_B, B0, nullptr, B0, E, DIM, DIM, 0);
  gemm(B0, Wm[4], P_C2_L1_B, nullptr, jx, B1, E, DIM, DIM, 1);
  gemm(B1, Wm[5], P_C2_L2_B, B0, nullptr, B0, E, DIM, DIM, 0);

  // ---- soft_agg over kk ----
  k_fill64<<<ew(HK_SIZE), blk, 0, stream>>>(hkK, ~0ull, HK_SIZE);
  k_fill32<<<ew(2), blk, 0, stream>>>((unsigned int*)counters, 0u, 2);
  k_hash_insert<<<ew(E), blk, 0, stream>>>(kk, 1ll, nullptr, hkK, nullptr, HK_SIZE - 1, E);
  k_group_assign<<<ew(HK_SIZE), blk, 0, stream>>>(hkK, hkI, counters, HK_SIZE, CAP_KK);
  k_group_get<<<ew(E), blk, 0, stream>>>(kk, 1ll, nullptr, hkK, hkI, HK_SIZE - 1, gidk, E);
  k_fill32<<<ew(CAP_KK * DIM), blk, 0, stream>>>((unsigned int*)GMk, NEG_INF_KEY, CAP_KK * DIM);
  k_fill32<<<ew(CAP_KK * DIM), blk, 0, stream>>>((unsigned int*)DNk, 0u, CAP_KK * DIM);
  k_fill32<<<ew(CAP_KK * DIM), blk, 0, stream>>>((unsigned int*)Yk, 0u, CAP_KK * DIM);
  gemm(B0, Wm[6], P_AKK_F_B, nullptr, nullptr, B1, E, DIM, DIM, 0);             // FX
  gemm(B0, Wm[7], P_AKK_G_B, nullptr, nullptr, B2, E, DIM, DIM, 0);             // GX
  k_segmax<<<ew(nE), blk, 0, stream>>>(B2, gidk, GMk, nE);
  k_gedens<<<ew(nE), blk, 0, stream>>>(B2, gidk, GMk, DNk, nE);                 // B2 := GE
  k_yacc<<<ew(nE), blk, 0, stream>>>(B1, B2, gidk, DNk, Yk, nE);
  gemm(Yk, Wm[8], P_AKK_H_B, nullptr, nullptr, Hk, CAP_KK, DIM, DIM, 0);
  k_scatadd<<<ew(nE), blk, 0, stream>>>(B0, Hk, gidk, nE);

  // ---- soft_agg over ii*12345+jj ----
  k_fill64<<<ew(HI_SIZE), blk, 0, stream>>>(hiK, ~0ull, HI_SIZE);
  k_hash_insert<<<ew(E), blk, 0, stream>>>(ii, 12345ll, jj, hiK, nullptr, HI_SIZE - 1, E);
  k_group_assign<<<ew(HI_SIZE), blk, 0, stream>>>(hiK, hiI, counters + 1, HI_SIZE, CAP_IJ);
  k_group_get<<<ew(E), blk, 0, stream>>>(ii, 12345ll, jj, hiK, hiI, HI_SIZE - 1, gidi, E);
  k_fill32<<<ew(CAP_IJ * DIM), blk, 0, stream>>>((unsigned int*)GMi, NEG_INF_KEY, CAP_IJ * DIM);
  k_fill32<<<ew(CAP_IJ * DIM), blk, 0, stream>>>((unsigned int*)DNi, 0u, CAP_IJ * DIM);
  k_fill32<<<ew(CAP_IJ * DIM), blk, 0, stream>>>((unsigned int*)Yi, 0u, CAP_IJ * DIM);
  gemm(B0, Wm[9],  P_AIJ_F_B, nullptr, nullptr, B1, E, DIM, DIM, 0);
  gemm(B0, Wm[10], P_AIJ_G_B, nullptr, nullptr, B2, E, DIM, DIM, 0);
  k_segmax<<<ew(nE), blk, 0, stream>>>(B2, gidi, GMi, nE);
  k_gedens<<<ew(nE), blk, 0, stream>>>(B2, gidi, GMi, DNi, nE);
  k_yacc<<<ew(nE), blk, 0, stream>>>(B1, B2, gidi, DNi, Yi, nE);
  gemm(Yi, Wm[11], P_AIJ_H_B, nullptr, nullptr, Hi, CAP_IJ, DIM, DIM, 0);
  k_scatadd<<<ew(nE), blk, 0, stream>>>(B0, Hi, gidi, nE);

  // ---- GRU block 1 ----
  k_ln<<<ew(E * 32), blk, 0, stream>>>(B0, (const float*)d_in[P_LN1_G],
      (const float*)d_in[P_LN1_B], B1, 0, E);                                   // L1
  gemm(B1, Wm[12], P_G1_GATE_B, nullptr, nullptr, B2, E, DIM, DIM, 2);          // gate
  gemm(B1, Wm[13], P_G1_R1_B,   nullptr, nullptr, B3, E, DIM, DIM, 1);          // relu(r1)
  gemm(B3, Wm[14], P_G1_R2_B,   nullptr, nullptr, B0, E, DIM, DIM, 0);          // r2
  k_fma3<<<ew(nE), blk, 0, stream>>>(B1, B2, B0, B0, nE);                       // X = L1 + gate*res

  // ---- GRU block 2 (final x written straight to d_out) ----
  k_ln<<<ew(E * 32), blk, 0, stream>>>(B0, (const float*)d_in[P_LN2_G],
      (const float*)d_in[P_LN2_B], B1, 0, E);
  gemm(B1, Wm[15], P_G2_GATE_B, nullptr, nullptr, B2, E, DIM, DIM, 2);
  gemm(B1, Wm[16], P_G2_R1_B,   nullptr, nullptr, B3, E, DIM, DIM, 1);
  gemm(B3, Wm[17], P_G2_R2_B,   nullptr, nullptr, B0, E, DIM, DIM, 0);
  k_fma3<<<ew(nE), blk, 0, stream>>>(B1, B2, B0, xo, nE);

  // ---- heads: d (linear) and w (sigmoid), gradient_clip = identity fwd ----
  k_heads<<<ew(E * 32), blk, 0, stream>>>(xo,
      (const float*)d_in[P_D_W], (const float*)d_in[P_D_B],
      (const float*)d_in[P_W_W], (const float*)d_in[P_W_B], dd, wo, E);
}
